// GraphFormer_32822140076797
// MI455X (gfx1250) — matrix operand order repaired and checked by reading, not compiled, not run on hardware
//
#include <hip/hip_runtime.h>
#include <hip/hip_bf16.h>
#include <math.h>

#define N_NODES 30000
#define N_EDGES 600000
#define DIM     256     // feature width at every stage
#define HEADS   4
#define LDS_PITCH 264   // 256 bf16 + 8 bf16 (16B) TDM pad per row -> bank spread

typedef __attribute__((ext_vector_type(16))) __bf16 v16bf;
typedef __attribute__((ext_vector_type(8)))  float  v8f;
typedef __attribute__((ext_vector_type(4)))  unsigned int v4u;
typedef __attribute__((ext_vector_type(8)))  int    v8i_t;
typedef __attribute__((ext_vector_type(4)))  int    v4i_t;

union BFrag { uint4 u[2]; v16bf v; };

// ---------------------------------------------------------------------------
// fp32 -> bf16 conversion (node features)
// ---------------------------------------------------------------------------
__global__ __launch_bounds__(256) void cvt_bf16_kernel(
    const float* __restrict__ src, __bf16* __restrict__ dst, int n) {
  int i = blockIdx.x * 256 + threadIdx.x;
  if (i < n) dst[i] = (__bf16)src[i];
}

// ---------------------------------------------------------------------------
// Pack weight W[k][n] (row-major, k-major) into Wt[n][k] bf16 (out-major)
// so each lane's B fragment is two contiguous 16B loads.
// ---------------------------------------------------------------------------
__global__ __launch_bounds__(256) void pack_w_kernel(
    const float* __restrict__ W, __bf16* __restrict__ Wt) {
  int n = blockIdx.x;      // 0..255 output channel
  int k = threadIdx.x;     // 0..255 reduction index
  Wt[(size_t)n * DIM + k] = (__bf16)W[(size_t)k * DIM + n];
}

// ---------------------------------------------------------------------------
// GEMM: C[30000,256] = A[30000,256](bf16) * W[256,256] + bias, via WMMA.
//
// One block owns a 32-row supertile. The A tile (32x256 bf16 = 16KB) is
// staged into LDS by the Tensor Data Mover (tensor_load_to_lds) with row
// padding (pad_interval=128 DWORDs, pad_amount=4 DWORDs -> 528B LDS pitch)
// so the 16 A-fragment lanes hit 16 distinct banks. Sync: s_wait_tensorcnt
// in the issuing wave + workgroup barrier.
//
// Wave w (of 8): M-tile = rows [16*(w>>2), +16), N-group = (w&3)*64.
// K-loop step 32: A frags from LDS (ds_load_b128), B frags from global
// (L2-hot 128KB weight), 4 accumulators -> 32 v_wmma per wave.
// ---------------------------------------------------------------------------
__global__ __launch_bounds__(256) void gemm_bf16_wmma(
    const __bf16* __restrict__ A,    // [30000,256] bf16 row-major
    const __bf16* __restrict__ Wt,   // [256(out),256(k)] bf16
    const float*  __restrict__ bias, // [256]
    float* __restrict__ C)           // [30000,256] fp32
{
  __shared__ __bf16 Abuf[32 * LDS_PITCH];   // 16.5KB (padded rows)

  const int row0 = blockIdx.x * 32;
  const int rows = min(32, N_NODES - row0);

  // --- TDM: stage A[row0 .. row0+rows) x 256 into LDS (wave 0 issues) ---
  if (threadIdx.x < 32) {
    unsigned long long ga = (unsigned long long)(uintptr_t)(A + (size_t)row0 * DIM);
    unsigned int lds_off  = (unsigned int)(uintptr_t)(&Abuf[0]);  // low32 = LDS byte offset

    v4u g0;
    g0[0] = 1u;                                   // count=1, user desc, no gather
    g0[1] = lds_off;                              // lds_addr
    g0[2] = (unsigned int)(ga & 0xffffffffu);     // global_addr[31:0]
    g0[3] = (unsigned int)((ga >> 32) & 0x01ffffffu) | (2u << 30); // [56:32], type=2

    v8i_t g1;
    g1[0] = (int)((1u << 16)        // data_size = 1 (2 bytes)
                | (1u << 20)        // pad_enable
                | (6u << 22)        // pad_interval: 128 DWORDs (one 512B row)
                | (3u << 25));      // pad_amount:   4 DWORDs (16B)
    g1[1] = (int)((DIM & 0xffffu) << 16);         // tensor_dim0[15:0]=256 (bits 63:48)
    g1[2] = (int)(((unsigned)rows & 0xffffu) << 16); // tensor_dim1[15:0]=rows (bits 95:80)
    g1[3] = (int)((DIM & 0xffffu) << 16);         // tile_dim0 = 256 (bits 127:112)
    g1[4] = (int)((unsigned)rows & 0xffffu);      // tile_dim1 = rows (bits 143:128)
    g1[5] = (int)DIM;                             // tensor_dim0_stride[31:0] = 256
    g1[6] = 0;                                    // stride hi, tensor_dim1_stride lo
    g1[7] = 0;

    v4i_t g2 = {};                                // unused for 2-D tile
    v4i_t g3 = {};
#if defined(__clang_major__) && (__clang_major__ >= 23)
    v8i_t g4 = {};
    __builtin_amdgcn_tensor_load_to_lds(g0, g1, g2, g3, g4, 0);
#else
    __builtin_amdgcn_tensor_load_to_lds(g0, g1, g2, g3, 0);
#endif
    __builtin_amdgcn_s_wait_tensorcnt(0);
  }
  __syncthreads();

  const int w     = threadIdx.x >> 5;   // wave 0..7
  const int lane  = threadIdx.x & 31;
  const int mloc  = (w >> 2) * 16;      // 0 or 16 within supertile
  const int nb    = (w & 3) * 64;
  const int m     = lane & 15;
  const int half  = lane >> 4;

  if (mloc >= rows) return;             // tail supertile: waves 4..7 idle (uniform)

  const __bf16* Alds = Abuf + (size_t)(mloc + m) * LDS_PITCH + half * 16;
  const __bf16* B0   = Wt + (size_t)(nb +  0 + m) * DIM + half * 16;
  const __bf16* B1   = Wt + (size_t)(nb + 16 + m) * DIM + half * 16;
  const __bf16* B2   = Wt + (size_t)(nb + 32 + m) * DIM + half * 16;
  const __bf16* B3   = Wt + (size_t)(nb + 48 + m) * DIM + half * 16;

  v8f acc0 = {}, acc1 = {}, acc2 = {}, acc3 = {};
#pragma unroll
  for (int kk = 0; kk < DIM; kk += 32) {
    BFrag a, b0, b1, b2, b3;
    a.u[0]  = *(const uint4*)(Alds + kk);
    a.u[1]  = *(const uint4*)(Alds + kk + 8);
    b0.u[0] = *(const uint4*)(B0 + kk);
    b0.u[1] = *(const uint4*)(B0 + kk + 8);
    b1.u[0] = *(const uint4*)(B1 + kk);
    b1.u[1] = *(const uint4*)(B1 + kk + 8);
    b2.u[0] = *(const uint4*)(B2 + kk);
    b2.u[1] = *(const uint4*)(B2 + kk + 8);
    b3.u[0] = *(const uint4*)(B3 + kk);
    b3.u[1] = *(const uint4*)(B3 + kk + 8);
    acc0 = __builtin_amdgcn_wmma_f32_16x16x32_bf16(false, a.v, false, b0.v,
                                                   (short)0, acc0, false, false);
    acc1 = __builtin_amdgcn_wmma_f32_16x16x32_bf16(false, a.v, false, b1.v,
                                                   (short)0, acc1, false, false);
    acc2 = __builtin_amdgcn_wmma_f32_16x16x32_bf16(false, a.v, false, b2.v,
                                                   (short)0, acc2, false, false);
    acc3 = __builtin_amdgcn_wmma_f32_16x16x32_bf16(false, a.v, false, b3.v,
                                                   (short)0, acc3, false, false);
  }

  const int   rbase = row0 + mloc + half * 8;
  const float bs0 = bias[nb +  0 + m];
  const float bs1 = bias[nb + 16 + m];
  const float bs2 = bias[nb + 32 + m];
  const float bs3 = bias[nb + 48 + m];
#pragma unroll
  for (int r = 0; r < 8; ++r) {
    float* crow = C + (size_t)(rbase + r) * DIM;
    crow[nb +  0 + m] = acc0[r] + bs0;
    crow[nb + 16 + m] = acc1[r] + bs1;
    crow[nb + 32 + m] = acc2[r] + bs2;
    crow[nb + 48 + m] = acc3[r] + bs3;
  }
}

// ---------------------------------------------------------------------------
// Init per-layer attention state: amax = -inf, denom = 0, BN accumulators = 0
// ---------------------------------------------------------------------------
__global__ __launch_bounds__(256) void init_attn_kernel(
    float* __restrict__ amax, float* __restrict__ denom,
    float* __restrict__ bn, int n4) {
  int i = blockIdx.x * 256 + threadIdx.x;
  if (i < n4) { amax[i] = -INFINITY; denom[i] = 0.0f; }
  if (i < 512) bn[i] = 0.0f;
}

// ---------------------------------------------------------------------------
// Pass 1: alpha[e,h] = dot(q[dst], k[src]) per head / 8 ; segment max (atomic)
// One wave32 per edge; lane l covers channels [8l, 8l+8) => head = l>>3,
// 8-lane butterfly reduction per head. q/k/v (92MB) are L2-resident.
// ---------------------------------------------------------------------------
__global__ __launch_bounds__(256) void edge_alpha_kernel(
    const int* __restrict__ ei, const float* __restrict__ Q,
    const float* __restrict__ K, float* __restrict__ alpha,
    float* __restrict__ amax) {
  int wid  = (blockIdx.x * 256 + threadIdx.x) >> 5;
  int lane = threadIdx.x & 31;
  if (wid >= N_EDGES) return;
  int src = ei[wid];
  int dst = ei[N_EDGES + wid];

  const float4* q = (const float4*)(Q + (size_t)dst * DIM) + lane * 2;
  const float4* k = (const float4*)(K + (size_t)src * DIM) + lane * 2;
  float4 q0 = q[0], q1 = q[1], k0 = k[0], k1 = k[1];
  float p = q0.x * k0.x + q0.y * k0.y + q0.z * k0.z + q0.w * k0.w +
            q1.x * k1.x + q1.y * k1.y + q1.z * k1.z + q1.w * k1.w;
  p += __shfl_xor(p, 1, 32);
  p += __shfl_xor(p, 2, 32);
  p += __shfl_xor(p, 4, 32);
  p *= 0.125f;                      // 1/sqrt(64)
  if ((lane & 7) == 0) {
    int h = lane >> 3;
    alpha[(size_t)wid * HEADS + h] = p;
    atomicMax(&amax[dst * HEADS + h], p);
  }
}

// ---------------------------------------------------------------------------
// Pass 2: ex = exp(alpha - amax[dst]); denom[dst] += ex (atomic)
// ---------------------------------------------------------------------------
__global__ __launch_bounds__(256) void edge_exp_kernel(
    const int* __restrict__ ei, float* __restrict__ alpha,
    const float* __restrict__ amax, float* __restrict__ denom) {
  int idx = blockIdx.x * 256 + threadIdx.x;
  if (idx >= N_EDGES * HEADS) return;
  int e = idx >> 2, h = idx & 3;
  int dst = ei[N_EDGES + e];
  float ex = __expf(alpha[idx] - amax[dst * HEADS + h]);
  alpha[idx] = ex;
  atomicAdd(&denom[dst * HEADS + h], ex);
}

// ---------------------------------------------------------------------------
// Pass 3: H[dst] += v[src] * (ex / denom[dst])  (atomic scatter-add)
// One wave32 per edge, lane covers 8 channels of one head.
// ---------------------------------------------------------------------------
__global__ __launch_bounds__(256) void edge_scatter_kernel(
    const int* __restrict__ ei, const float* __restrict__ V,
    const float* __restrict__ alpha, const float* __restrict__ denom,
    float* __restrict__ H) {
  int wid  = (blockIdx.x * 256 + threadIdx.x) >> 5;
  int lane = threadIdx.x & 31;
  if (wid >= N_EDGES) return;
  int src = ei[wid];
  int dst = ei[N_EDGES + wid];
  int h   = lane >> 3;
  float w = alpha[(size_t)wid * HEADS + h] / denom[dst * HEADS + h];

  const float4* v = (const float4*)(V + (size_t)src * DIM) + lane * 2;
  float4 v0 = v[0], v1 = v[1];
  float* o = H + (size_t)dst * DIM + lane * 8;
  atomicAdd(o + 0, v0.x * w);
  atomicAdd(o + 1, v0.y * w);
  atomicAdd(o + 2, v0.z * w);
  atomicAdd(o + 3, v0.w * w);
  atomicAdd(o + 4, v1.x * w);
  atomicAdd(o + 5, v1.y * w);
  atomicAdd(o + 6, v1.z * w);
  atomicAdd(o + 7, v1.w * w);
}

// ---------------------------------------------------------------------------
// BatchNorm statistics: per-channel sum / sumsq (thread == channel)
// ---------------------------------------------------------------------------
__global__ __launch_bounds__(256) void bn_stats_kernel(
    const float* __restrict__ H, float* __restrict__ bn) {
  int c  = threadIdx.x;
  int r0 = blockIdx.x * 128;
  int r1 = min(r0 + 128, N_NODES);
  float s = 0.0f, ss = 0.0f;
  for (int r = r0; r < r1; ++r) {
    float v = H[(size_t)r * DIM + c];
    s += v; ss += v * v;
  }
  atomicAdd(&bn[c], s);
  atomicAdd(&bn[DIM + c], ss);
}

// bn[0..255]=sum, bn[256..511]=sumsq -> bn[512..767]=scale, bn[768..1023]=shift
__global__ __launch_bounds__(256) void bn_finalize_kernel(
    float* __restrict__ bn, const float* __restrict__ g,
    const float* __restrict__ be) {
  int c = threadIdx.x;
  const float inv_n = 1.0f / (float)N_NODES;
  float mean = bn[c] * inv_n;
  float var  = bn[DIM + c] * inv_n - mean * mean;
  float sc   = g[c] * rsqrtf(var + 1e-5f);
  bn[2 * DIM + c] = sc;
  bn[3 * DIM + c] = be[c] - mean * sc;
}

// Apply BN + leaky ReLU, emit bf16 features for the next layer's GEMMs
__global__ __launch_bounds__(256) void bn_apply_bf16_kernel(
    const float* __restrict__ H, const float* __restrict__ bn,
    __bf16* __restrict__ out, int n) {
  int i = blockIdx.x * 256 + threadIdx.x;
  if (i >= n) return;
  int c = i & (DIM - 1);
  float y = H[i] * bn[2 * DIM + c] + bn[3 * DIM + c];
  y = (y >= 0.0f) ? y : 0.1f * y;
  out[i] = (__bf16)y;
}

// Apply BN + leaky ReLU in place (final fp32 output)
__global__ __launch_bounds__(256) void bn_apply_f32_kernel(
    float* __restrict__ H, const float* __restrict__ bn, int n) {
  int i = blockIdx.x * 256 + threadIdx.x;
  if (i >= n) return;
  int c = i & (DIM - 1);
  float y = H[i] * bn[2 * DIM + c] + bn[3 * DIM + c];
  H[i] = (y >= 0.0f) ? y : 0.1f * y;
}

// ---------------------------------------------------------------------------
// Host-side orchestration
// ---------------------------------------------------------------------------
extern "C" void kernel_launch(void* const* d_in, const int* in_sizes, int n_in,
                              void* d_out, int out_size, void* d_ws, size_t ws_size,
                              hipStream_t stream) {
  (void)in_sizes; (void)n_in; (void)out_size; (void)ws_size;
  const float* x  = (const float*)d_in[0];
  const int*   ei = (const int*)d_in[1];
  const float* W1[4] = {(const float*)d_in[2], (const float*)d_in[4],
                        (const float*)d_in[6], (const float*)d_in[8]};   // q,k,v,s
  const float* B1[4] = {(const float*)d_in[3], (const float*)d_in[5],
                        (const float*)d_in[7], (const float*)d_in[9]};
  const float* g1  = (const float*)d_in[10];
  const float* be1 = (const float*)d_in[11];
  const float* W2[4] = {(const float*)d_in[12], (const float*)d_in[14],
                        (const float*)d_in[16], (const float*)d_in[18]};
  const float* B2[4] = {(const float*)d_in[13], (const float*)d_in[15],
                        (const float*)d_in[17], (const float*)d_in[19]};
  const float* g2  = (const float*)d_in[20];
  const float* be2 = (const float*)d_in[21];

  // Workspace carve-up (256B aligned), ~149MB total
  char* ws = (char*)d_ws;
  size_t off = 0;
  auto carve = [&](size_t bytes) -> void* {
    off = (off + 255) & ~(size_t)255;
    void* p = ws + off;
    off += bytes;
    return p;
  };
  __bf16* xbf   = (__bf16*)carve((size_t)N_NODES * DIM * sizeof(__bf16));
  float*  Qb    = (float*) carve((size_t)N_NODES * DIM * sizeof(float));
  float*  Kb    = (float*) carve((size_t)N_NODES * DIM * sizeof(float));
  float*  Vb    = (float*) carve((size_t)N_NODES * DIM * sizeof(float));
  float*  H1    = (float*) carve((size_t)N_NODES * DIM * sizeof(float));
  float*  alpha = (float*) carve((size_t)N_EDGES * HEADS * sizeof(float));
  float*  amax  = (float*) carve((size_t)N_NODES * HEADS * sizeof(float));
  float*  denom = (float*) carve((size_t)N_NODES * HEADS * sizeof(float));
  __bf16* Wt    = (__bf16*)carve((size_t)DIM * DIM * sizeof(__bf16));
  float*  bn    = (float*) carve(4 * DIM * sizeof(float));

  const int NFEAT      = N_NODES * DIM;                 // 7,680,000
  const int GEMM_GRID  = (N_NODES + 31) / 32;           // 938 (32-row supertiles)
  const int EDGE_GRID  = N_EDGES / 8;                   // 75,000 (wave per edge)
  const int EXP_GRID   = (N_EDGES * HEADS) / 256;       // 9,375
  const int INIT_GRID  = (N_NODES * HEADS + 255) / 256; // 469
  const int STAT_GRID  = (N_NODES + 127) / 128;         // 235
  const int ELT_GRID   = NFEAT / 256;                   // 30,000

  // bf16 input features
  cvt_bf16_kernel<<<ELT_GRID, 256, 0, stream>>>(x, xbf, NFEAT);

  auto run_layer = [&](const __bf16* Abf, const float* const* W,
                       const float* const* B, const float* g, const float* be,
                       float* H, __bf16* next_bf) {
    // skip connection GEMM initializes the accumulator: H = A @ Ws + bs
    pack_w_kernel<<<DIM, DIM, 0, stream>>>(W[3], Wt);
    gemm_bf16_wmma<<<GEMM_GRID, 256, 0, stream>>>(Abf, Wt, B[3], H);
    pack_w_kernel<<<DIM, DIM, 0, stream>>>(W[0], Wt);
    gemm_bf16_wmma<<<GEMM_GRID, 256, 0, stream>>>(Abf, Wt, B[0], Qb);
    pack_w_kernel<<<DIM, DIM, 0, stream>>>(W[1], Wt);
    gemm_bf16_wmma<<<GEMM_GRID, 256, 0, stream>>>(Abf, Wt, B[1], Kb);
    pack_w_kernel<<<DIM, DIM, 0, stream>>>(W[2], Wt);
    gemm_bf16_wmma<<<GEMM_GRID, 256, 0, stream>>>(Abf, Wt, B[2], Vb);

    init_attn_kernel<<<INIT_GRID, 256, 0, stream>>>(amax, denom, bn,
                                                    N_NODES * HEADS);
    edge_alpha_kernel<<<EDGE_GRID, 256, 0, stream>>>(ei, Qb, Kb, alpha, amax);
    edge_exp_kernel<<<EXP_GRID, 256, 0, stream>>>(ei, alpha, amax, denom);
    edge_scatter_kernel<<<EDGE_GRID, 256, 0, stream>>>(ei, Vb, alpha, denom, H);

    bn_stats_kernel<<<STAT_GRID, 256, 0, stream>>>(H, bn);
    bn_finalize_kernel<<<1, 256, 0, stream>>>(bn, g, be);
    if (next_bf)
      bn_apply_bf16_kernel<<<ELT_GRID, 256, 0, stream>>>(H, bn, next_bf, NFEAT);
    else
      bn_apply_f32_kernel<<<ELT_GRID, 256, 0, stream>>>(H, bn, NFEAT);
  };

  // Layer 1: output features re-use xbf (x already consumed by GEMMs)
  run_layer(xbf, W1, B1, g1, be1, H1, xbf);
  // Layer 2: accumulate directly into d_out, finalize in place
  run_layer(xbf, W2, B2, g2, be2, (float*)d_out, nullptr);
}
